// Transformer_20822001451066
// MI455X (gfx1250) — matrix operand order repaired and checked by reading, not compiled, not run on hardware
//
#include <hip/hip_runtime.h>
#include <hip/hip_bf16.h>
#include <stdint.h>

// ---------------------------------------------------------------------------
// Transformer forward (B=4, S=2048, D=1024, DFF=4096, L=8) for gfx1250.
// All GEMM operands are bf16 in global memory ([M][K] A, [N][K] B); tiles are
// staged into double-buffered LDS via GLOBAL_LOAD_ASYNC_TO_LDS_B128 (ASYNCcnt
// path). Compute is v_wmma_f32_16x16x32_bf16 with f32 accumulate. ~230 MB ws.
// ---------------------------------------------------------------------------

#define DM   1024
#define DFFC 4096
#define LC   8
#define BC   4
#define SC   2048
#define NTOK (BC * SC)          // 8192 rows
#define EPSC 1e-5f

typedef __attribute__((ext_vector_type(8)))  float          v8f;
typedef __attribute__((ext_vector_type(16))) __bf16         v16bf;
typedef __attribute__((ext_vector_type(8)))  unsigned short ushort8;
typedef int v4i __attribute__((vector_size(16)));

union FragU {
    ushort8 h[2];
    v16bf   b;
};

__device__ __forceinline__ unsigned short f2bf(float x) {
    unsigned int u = __float_as_uint(x);
    u += 0x7FFFu + ((u >> 16) & 1u);      // round-to-nearest-even
    return (unsigned short)(u >> 16);
}

// ---- CDNA5 async global->LDS copy (guarded; sync b128 fallback) ----
#if defined(__has_builtin)
#  if __has_builtin(__builtin_amdgcn_global_load_async_to_lds_b128)
#    define USE_ASYNC_LDS 1
#  endif
#endif

#ifdef USE_ASYNC_LDS
typedef __attribute__((address_space(1))) v4i* gv4i_p;   // global int4*
typedef __attribute__((address_space(3))) v4i* lv4i_p;   // LDS int4*
__device__ __forceinline__ void copy16_g2l(const unsigned short* gsrc,
                                           unsigned short* ldst) {
    __builtin_amdgcn_global_load_async_to_lds_b128(
        (gv4i_p)(uintptr_t)gsrc, (lv4i_p)(uintptr_t)ldst, 0, 0);
}
__device__ __forceinline__ void wait_async() {
#  if __has_builtin(__builtin_amdgcn_s_wait_asynccnt)
    __builtin_amdgcn_s_wait_asynccnt(0);
#  else
    asm volatile("s_wait_asynccnt 0" ::: "memory");
#  endif
}
#else
__device__ __forceinline__ void copy16_g2l(const unsigned short* gsrc,
                                           unsigned short* ldst) {
    *(ushort8*)ldst = *(const ushort8*)gsrc;
}
__device__ __forceinline__ void wait_async() {}
#endif

// Epilogue output modes
#define OUT_F32     0   // Cf[row*ldc+col]  = v
#define OUT_F32_ADD 1   // Cf[row*ldc+col] += v
#define OUT_BF16    2   // Cb[row*ldc+col]  = bf16(v)
#define OUT_BF16_T  3   // Cb[col*ldc+row]  = bf16(v)   (transposed store)

// ---------------------------------------------------------------------------
// Tiled bf16-WMMA GEMM:  C (+)= alpha * A @ B^T_layout + bias
//   A: bf16 [M][K] row-major (lda); B: bf16 [N][K] row-major (ldb)
//   Cf (f32) or Cb (bf16) per outMode; ldc is C's leading dim (rows for
//   OUT_BF16_T). gridDim.z batches via element strides.
//   Block: 128 threads = 4 waves; tile 128x128; wave = 64x64 = 4x4 WMMAs.
//   K-step 32, double-buffered LDS, one barrier per chunk; staging of chunk
//   i+1 (async DMA) overlaps the 16 WMMAs of chunk i.
// Requires: M%128==0, N%128==0, K%32==0.
// ---------------------------------------------------------------------------
#define AROW 40   // LDS row stride (bf16 elems): 80B, keeps 16B alignment
#define BROW 40

__global__ __launch_bounds__(128) void gemm_bf16_wmma(
    const unsigned short* __restrict__ A, const unsigned short* __restrict__ B,
    const float* __restrict__ bias,
    float* __restrict__ Cf, unsigned short* __restrict__ Cb,
    int M, int N, int K, int lda, int ldb, int ldc,
    long long strideA, long long strideB, long long strideC,
    float alpha, int doRelu, int outMode)
{
    __shared__ unsigned short Abf[2][128 * AROW];   // [r*AROW + k], k in [0,32)
    __shared__ unsigned short Bbf[2][128 * BROW];   // [n*BROW + k], k in [0,32)

    const long long bz = blockIdx.z;
    A += bz * strideA;
    B += bz * strideB;
    if (Cf) Cf += bz * strideC;
    if (Cb) Cb += bz * strideC;

    const int tid  = threadIdx.x;
    const int lane = tid & 31;
    const int wave = tid >> 5;
    const int m0   = blockIdx.y * 128;
    const int n0   = blockIdx.x * 128;
    const int wr   = (wave >> 1) * 64;    // wave's row offset inside tile
    const int wc   = (wave & 1) * 64;     // wave's col offset inside tile
    const int l15  = lane & 15;
    const int khi  = lane >> 4;

    v8f acc[4][4] = {};

    // Stage one 128x32 bf16 tile pair: 512 chunks of 16B each, 4 per thread.
    auto stage = [&](int buf, int k0) {
        #pragma unroll
        for (int i = 0; i < 4; i++) {
            int c  = i * 128 + tid;
            int r  = c >> 2;              // row (A) / row-of-B[N][K] (B)
            int o8 = (c & 3) << 3;        // 8-elem offset within 32-k row
            copy16_g2l(&A[(long long)(m0 + r) * lda + (k0 + o8)],
                       &Abf[buf][r * AROW + o8]);
            copy16_g2l(&B[(long long)(n0 + r) * ldb + (k0 + o8)],
                       &Bbf[buf][r * BROW + o8]);
        }
    };

    stage(0, 0);

    for (int k0 = 0; k0 < K; k0 += 32) {
        wait_async();                          // our async copies done
        __syncthreads();                       // everyone's copies visible
        const int buf = (k0 >> 5) & 1;

        if (k0 + 32 < K) {
            if (k0 + 64 < K) {                 // L2 warm 2 chunks ahead
                __builtin_prefetch(&A[(long long)(m0 + wave * 32) * lda + k0 + 64], 0, 1);
                __builtin_prefetch(&B[(long long)(n0 + wave * 32) * ldb + k0 + 64], 0, 1);
            }
            stage(buf ^ 1, k0 + 32);           // overlaps WMMAs below
        }

        // ---- fragments ----
        // A 16x32 bf16: lane (l15,khi) of row M holds K {khi*8+e, 16+khi*8+e}
        FragU afr[4], bfr[4];
        #pragma unroll
        for (int mi = 0; mi < 4; mi++) {
            const unsigned short* p = &Abf[buf][(wr + mi * 16 + l15) * AROW];
            afr[mi].h[0] = *(const ushort8*)(p + khi * 8);
            afr[mi].h[1] = *(const ushort8*)(p + 16 + khi * 8);
        }
        // B 32x16 bf16: lane holds col n=l15, K {khi*8+e, 16+khi*8+e}
        #pragma unroll
        for (int ni = 0; ni < 4; ni++) {
            const unsigned short* p = &Bbf[buf][(wc + ni * 16 + l15) * BROW + khi * 8];
            bfr[ni].h[0] = *(const ushort8*)(p);
            bfr[ni].h[1] = *(const ushort8*)(p + 16);
        }

        // ---- 16 WMMAs: 64x64 per wave ----
        #pragma unroll
        for (int mi = 0; mi < 4; mi++)
            #pragma unroll
            for (int ni = 0; ni < 4; ni++)
                acc[mi][ni] = __builtin_amdgcn_wmma_f32_16x16x32_bf16(
                    false, afr[mi].b, false, bfr[ni].b,
                    (short)0, acc[mi][ni], false, false);
    }

    // ---- epilogue: C/D layout M = i + (lane>>4)*8, N = lane&15 ----
    #pragma unroll
    for (int mi = 0; mi < 4; mi++) {
        #pragma unroll
        for (int ni = 0; ni < 4; ni++) {
            int col = n0 + wc + ni * 16 + l15;
            float bb = bias ? bias[col] : 0.0f;
            int rowb = m0 + wr + mi * 16 + khi * 8;
            if (outMode == OUT_BF16_T) {
                ushort8 s;
                #pragma unroll
                for (int i = 0; i < 8; i++) {
                    float v = acc[mi][ni][i] * alpha + bb;
                    if (doRelu) v = fmaxf(v, 0.0f);
                    s[i] = f2bf(v);
                }
                *(ushort8*)&Cb[(long long)col * ldc + rowb] = s;  // 16B store
            } else {
                #pragma unroll
                for (int i = 0; i < 8; i++) {
                    float v = acc[mi][ni][i] * alpha + bb;
                    if (doRelu) v = fmaxf(v, 0.0f);
                    long long off = (long long)(rowb + i) * ldc + col;
                    if      (outMode == OUT_F32)     Cf[off]  = v;
                    else if (outMode == OUT_F32_ADD) Cf[off] += v;
                    else                             Cb[off]  = f2bf(v);
                }
            }
        }
    }
}

// ---------------------------------------------------------------------------
// Fused convert + transpose: W f32 [K][N] -> WT bf16 [N][K]
// block (32,8), 32x32 tile via LDS, coalesced both sides
// ---------------------------------------------------------------------------
__global__ __launch_bounds__(256) void cvtT_k(
    const float* __restrict__ W, unsigned short* __restrict__ WT, int K, int N)
{
    __shared__ float t[32][33];
    int n0 = blockIdx.x * 32, k0 = blockIdx.y * 32;
    int tx = threadIdx.x, ty = threadIdx.y;
    #pragma unroll
    for (int j = 0; j < 4; j++)
        t[ty + j * 8][tx] = W[(long long)(k0 + ty + j * 8) * N + (n0 + tx)];
    __syncthreads();
    #pragma unroll
    for (int j = 0; j < 4; j++)
        WT[(long long)(n0 + ty + j * 8) * K + (k0 + tx)] = f2bf(t[tx][ty + j * 8]);
}

// ---------------------------------------------------------------------------
// Embedding + sinusoidal positional encoding (f32 residual stream)
// ---------------------------------------------------------------------------
__global__ __launch_bounds__(256) void embed_pe_k(
    const int* __restrict__ tokens, const float* __restrict__ emb,
    float* __restrict__ x)
{
    long long idx = (long long)blockIdx.x * blockDim.x + threadIdx.x;
    if (idx >= (long long)NTOK * DM) return;
    int d = (int)(idx & (DM - 1));
    int t = (int)(idx >> 10);             // flat (b*S + s)
    int s = t & (SC - 1);
    int tok = tokens[t];
    float freq = __expf(-(float)(d & ~1) * (9.210340371976184f / (float)DM));
    float ang  = (float)s * freq;
    float pe   = (d & 1) ? __cosf(ang) : __sinf(ang);
    x[idx] = emb[(long long)tok * DM + d] + pe;
}

// ---------------------------------------------------------------------------
// LayerNorm: one 256-thread block per row of 1024; bf16 output
// ---------------------------------------------------------------------------
__global__ __launch_bounds__(256) void layernorm_k(
    const float* __restrict__ x, const float* __restrict__ g,
    const float* __restrict__ b, unsigned short* __restrict__ out)
{
    __shared__ float red[256];
    const int row = blockIdx.x;
    const int tid = threadIdx.x;
    const float* xr = x + (long long)row * DM;
    float v[4];
    float s = 0.0f;
    #pragma unroll
    for (int i = 0; i < 4; i++) { v[i] = xr[tid + i * 256]; s += v[i]; }
    red[tid] = s; __syncthreads();
    for (int o = 128; o > 0; o >>= 1) {
        if (tid < o) red[tid] += red[tid + o];
        __syncthreads();
    }
    float mu = red[0] * (1.0f / DM);
    __syncthreads();
    s = 0.0f;
    #pragma unroll
    for (int i = 0; i < 4; i++) { float d = v[i] - mu; s += d * d; }
    red[tid] = s; __syncthreads();
    for (int o = 128; o > 0; o >>= 1) {
        if (tid < o) red[tid] += red[tid + o];
        __syncthreads();
    }
    float rstd = __frsqrt_rn(red[0] * (1.0f / DM) + EPSC);
    unsigned short* orow = out + (long long)row * DM;
    #pragma unroll
    for (int i = 0; i < 4; i++) {
        int c = tid + i * 256;
        orow[c] = f2bf((v[i] - mu) * rstd * g[c] + b[c]);
    }
}

// ---------------------------------------------------------------------------
// Softmax over rows of length 2048: f32 in, bf16 out
// ---------------------------------------------------------------------------
__global__ __launch_bounds__(256) void softmax_k(
    const float* __restrict__ scores, unsigned short* __restrict__ attn)
{
    __shared__ float red[256];
    const int tid = threadIdx.x;
    const float* r = scores + (long long)blockIdx.x * SC;
    unsigned short* o = attn + (long long)blockIdx.x * SC;
    float v[8];
    float mx = -3.402823466e+38f;
    #pragma unroll
    for (int i = 0; i < 8; i++) { v[i] = r[tid + i * 256]; mx = fmaxf(mx, v[i]); }
    red[tid] = mx; __syncthreads();
    for (int w = 128; w > 0; w >>= 1) {
        if (tid < w) red[tid] = fmaxf(red[tid], red[tid + w]);
        __syncthreads();
    }
    mx = red[0]; __syncthreads();
    float s = 0.0f;
    #pragma unroll
    for (int i = 0; i < 8; i++) { v[i] = __expf(v[i] - mx); s += v[i]; }
    red[tid] = s; __syncthreads();
    for (int w = 128; w > 0; w >>= 1) {
        if (tid < w) red[tid] += red[tid + w];
        __syncthreads();
    }
    float inv = 1.0f / red[0];
    #pragma unroll
    for (int i = 0; i < 8; i++) o[tid + i * 256] = f2bf(v[i] * inv);
}

// ---------------------------------------------------------------------------
// Host-side orchestration
// ---------------------------------------------------------------------------
extern "C" void kernel_launch(void* const* d_in, const int* in_sizes, int n_in,
                              void* d_out, int out_size, void* d_ws, size_t ws_size,
                              hipStream_t stream) {
    const int*   tokens = (const int*)  d_in[0];
    const float* emb    = (const float*)d_in[1];
    const float* Wq     = (const float*)d_in[2];
    const float* bq     = (const float*)d_in[3];
    const float* Wk     = (const float*)d_in[4];
    const float* bk     = (const float*)d_in[5];
    const float* Wv     = (const float*)d_in[6];
    const float* bv     = (const float*)d_in[7];
    const float* g1     = (const float*)d_in[8];
    const float* beta1  = (const float*)d_in[9];
    const float* g2     = (const float*)d_in[10];
    const float* beta2  = (const float*)d_in[11];
    const float* W1     = (const float*)d_in[12];
    const float* bm1    = (const float*)d_in[13];
    const float* W2     = (const float*)d_in[14];
    const float* bm2    = (const float*)d_in[15];

    // ---- workspace layout: 24M floats + 67M ushorts = 230 MB ----
    const size_t XN = (size_t)NTOK * DM;            // 8M
    float* x      = (float*)d_ws;                   // f32 residual stream
    float* scores = x + XN;                         // 16M f32
    unsigned short* us  = (unsigned short*)(scores + (size_t)BC * SC * SC);
    unsigned short* hbf = us;                       // LN output       8M
    unsigned short* qbf = hbf + XN;                 //                 8M
    unsigned short* kbf = qbf + XN;                 //                 8M
    unsigned short* vT  = kbf + XN;                 // [D][B*S]        8M
    unsigned short* att = vT + XN;                  // softmax out    16M
    unsigned short* ffh = att + (size_t)BC * SC * SC;   // 2048x4096   8M
    unsigned short* WqT = ffh + (size_t)SC * DFFC;  // [N][K] weights
    unsigned short* WkT = WqT + (size_t)DM * DM;
    unsigned short* WvT = WkT + (size_t)DM * DM;
    unsigned short* W1T = WvT + (size_t)DM * DM;    // [4096][1024]
    unsigned short* W2T = W1T + (size_t)DM * DFFC;  // [1024][4096]

    const dim3 blk(128);
    const float inv_sqrt_d = 1.0f / 32.0f;          // 1/sqrt(1024)

    embed_pe_k<<<(unsigned)(XN / 256), 256, 0, stream>>>(tokens, emb, x);

    for (int l = 0; l < LC; l++) {
        const float* bqL = bq + (size_t)l * DM;
        const float* bkL = bk + (size_t)l * DM;
        const float* bvL = bv + (size_t)l * DM;
        const float* b1L = bm1 + (size_t)l * DFFC;
        const float* b2L = bm2 + (size_t)l * DM;

        // ---- convert + transpose this layer's weights to bf16 [N][K] ----
        dim3 ctb(32, 8);
        cvtT_k<<<dim3(DM / 32, DM / 32), ctb, 0, stream>>>(Wq + (size_t)l * DM * DM, WqT, DM, DM);
        cvtT_k<<<dim3(DM / 32, DM / 32), ctb, 0, stream>>>(Wk + (size_t)l * DM * DM, WkT, DM, DM);
        cvtT_k<<<dim3(DM / 32, DM / 32), ctb, 0, stream>>>(Wv + (size_t)l * DM * DM, WvT, DM, DM);
        cvtT_k<<<dim3(DFFC / 32, DM / 32), ctb, 0, stream>>>(W1 + (size_t)l * DM * DFFC, W1T, DM, DFFC);
        cvtT_k<<<dim3(DM / 32, DFFC / 32), ctb, 0, stream>>>(W2 + (size_t)l * DFFC * DM, W2T, DFFC, DM);

        // LN1: x -> hbf
        layernorm_k<<<NTOK, 256, 0, stream>>>(x, g1 + (size_t)l * DM, beta1 + (size_t)l * DM, hbf);

        // Q,K,V projections: [8192x1024] @ [1024x1024]^T-layout + bias
        dim3 gqkv(DM / 128, NTOK / 128, 1);
        gemm_bf16_wmma<<<gqkv, blk, 0, stream>>>(hbf, WqT, bqL, nullptr, qbf,
                                                 NTOK, DM, DM, DM, DM, DM,
                                                 0, 0, 0, 1.0f, 0, OUT_BF16);
        gemm_bf16_wmma<<<gqkv, blk, 0, stream>>>(hbf, WkT, bkL, nullptr, kbf,
                                                 NTOK, DM, DM, DM, DM, DM,
                                                 0, 0, 0, 1.0f, 0, OUT_BF16);
        // V written transposed: vT[d][b*S+t]  (ldc = NTOK rows)
        gemm_bf16_wmma<<<gqkv, blk, 0, stream>>>(hbf, WvT, bvL, nullptr, vT,
                                                 NTOK, DM, DM, DM, DM, NTOK,
                                                 0, 0, 0, 1.0f, 0, OUT_BF16_T);

        // scores = Q @ K^T / sqrt(D): A=q [S][D], B=k [S][D] = [N][K]
        dim3 gsc(SC / 128, SC / 128, BC);
        gemm_bf16_wmma<<<gsc, blk, 0, stream>>>(qbf, kbf, nullptr, scores, nullptr,
                                                SC, SC, DM, DM, DM, SC,
                                                (long long)SC * DM, (long long)SC * DM,
                                                (long long)SC * SC,
                                                inv_sqrt_d, 0, OUT_F32);

        // softmax (f32 -> bf16 attn)
        softmax_k<<<BC * SC, 256, 0, stream>>>(scores, att);

        // x += attn @ V: A=att [S][S], B=vT + b*S with ldb = NTOK  ([N=d][K=t])
        dim3 gav(DM / 128, SC / 128, BC);
        gemm_bf16_wmma<<<gav, blk, 0, stream>>>(att, vT, nullptr, x, nullptr,
                                                SC, DM, SC, SC, NTOK, DM,
                                                (long long)SC * SC, (long long)SC,
                                                (long long)SC * DM,
                                                1.0f, 0, OUT_F32_ADD);

        // LN2: x -> hbf
        layernorm_k<<<NTOK, 256, 0, stream>>>(x, g2 + (size_t)l * DM, beta2 + (size_t)l * DM, hbf);

        // MLP, chunked over 2048-row slices to bound the hidden buffer
        for (int c = 0; c < NTOK / SC; c++) {
            const unsigned short* hA = hbf + (size_t)c * SC * DM;
            float*                xC = x   + (size_t)c * SC * DM;
            dim3 gm1(DFFC / 128, SC / 128, 1);
            gemm_bf16_wmma<<<gm1, blk, 0, stream>>>(hA, W1T, b1L, nullptr, ffh,
                                                    SC, DFFC, DM, DM, DM, DFFC,
                                                    0, 0, 0, 1.0f, 1 /*relu*/, OUT_BF16);
            dim3 gm2(DM / 128, SC / 128, 1);
            gemm_bf16_wmma<<<gm2, blk, 0, stream>>>(ffh, W2T, b2L, xC, nullptr,
                                                    SC, DM, DFFC, DFFC, DFFC, DM,
                                                    0, 0, 0, 1.0f, 0, OUT_F32_ADD);
        }
    }

    (void)hipMemcpyAsync(d_out, x, (size_t)out_size * sizeof(float),
                         hipMemcpyDeviceToDevice, stream);
}
